// DiffAttn_25297357374086
// MI455X (gfx1250) — matrix operand order repaired and checked by reading, not compiled, not run on hardware
//
#include <hip/hip_runtime.h>

// Problem constants (from reference)
#define B_   2
#define N_   4096
#define D_   768
#define DK_  64
#define DV_  64
#define NSPLIT 4
#define CHUNK (N_ / NSPLIT)          // 1024 keys per chunk
// scale = 1/sqrt(BLOCK_SIZE=4096) = 1/64
#define SSCALE 0.015625f

typedef __attribute__((ext_vector_type(16))) __bf16 bf16x16;
typedef __attribute__((ext_vector_type(8)))  float  v8f;
typedef __attribute__((ext_vector_type(4)))  int    v4i;
typedef __attribute__((address_space(1))) v4i gv4i;  // global-space int4
typedef __attribute__((address_space(3))) v4i lv4i;  // LDS-space int4

union Frag {
    bf16x16 f;
    uint4   q[2];
};

// A-matrix fragment (16x32 bf16, row-major source, stride in elements).
// lanes 0-15: row=lane,   K in {0..7} U {16..23}
// lanes16-31: row=lane-16,K in {8..15} U {24..31}
__device__ __forceinline__ void loadA(Frag& fr, const __bf16* p, int stride, int lane) {
    const int r = lane & 15, h = lane >> 4;
    fr.q[0] = *(const uint4*)(p + (size_t)r * stride + h * 8);
    fr.q[1] = *(const uint4*)(p + (size_t)r * stride + 16 + h * 8);
}

// B-matrix fragment (32x16 bf16). Source is B^T row-major: row r = output column,
// K contiguous. lanes 0-15 hold K in {0..7} U {16..23}, lanes 16-31 hold K in {8..15} U {24..31}.
__device__ __forceinline__ void loadB(Frag& fr, const __bf16* p, int stride, int lane) {
    const int r = lane & 15, h = lane >> 4;
    fr.q[0] = *(const uint4*)(p + (size_t)r * stride + h * 8);
    fr.q[1] = *(const uint4*)(p + (size_t)r * stride + 16 + h * 8);
}

__device__ __forceinline__ v8f wmma_bf16(const Frag& a, const Frag& b, v8f c) {
    return __builtin_amdgcn_wmma_f32_16x16x32_bf16(
        false, a.f, false, b.f, (short)0, c, false, false);
}

// 16-byte global -> LDS copy; async (ASYNCcnt) when the builtin exists.
__device__ __forceinline__ void async_copy16(const void* gsrc, void* ldst) {
#if defined(__has_builtin) && __has_builtin(__builtin_amdgcn_global_load_async_to_lds_b128)
    __builtin_amdgcn_global_load_async_to_lds_b128((gv4i*)gsrc, (lv4i*)ldst, 0, 0);
#else
    *(uint4*)ldst = *(const uint4*)gsrc;
#endif
}
__device__ __forceinline__ void wait_async() {
    asm volatile("s_wait_asynccnt 0x0" ::: "memory");
}
__device__ __forceinline__ void wait_lds() {
    asm volatile("s_wait_dscnt 0x0" ::: "memory");
}

// ---------------- f32 -> bf16 conversion ----------------
__global__ void cvt_f32_bf16(const float* __restrict__ src, __bf16* __restrict__ dst, int count) {
    int i = blockIdx.x * blockDim.x + threadIdx.x;
    if (i < count) dst[i] = (__bf16)src[i];
}

// ---------------- fused QKV projection ----------------
// One wave computes 16 rows x all 192 output cols (q|k|v). The 192x32 weight
// K-slice is async-staged into LDS once per block (8 waves share it).
__global__ __launch_bounds__(256) void qkv_proj(
    const __bf16* __restrict__ xb, const __bf16* __restrict__ Wb,
    const float* __restrict__ bq, const float* __restrict__ bk, const float* __restrict__ bv,
    __bf16* __restrict__ qb, __bf16* __restrict__ kb, __bf16* __restrict__ vT)
{
    __shared__ __align__(16) __bf16 Wl[192 * 40];   // padded stride 40 (bank-conflict free)

    const int tid  = threadIdx.x;
    const int lane = tid & 31, warp = tid >> 5;
    const int rt   = blockIdx.x * 8 + warp;          // 0 .. B*(N/16)-1
    const int b    = rt / (N_ / 16);
    const int row0 = (rt % (N_ / 16)) * 16;
    const int r = lane & 15, h = lane >> 4;

    const __bf16* ap = xb + ((size_t)(b * N_ + row0)) * D_;

    v8f acc[12];
    v8f vzero = {};
#pragma unroll
    for (int t = 0; t < 12; t++) acc[t] = vzero;

    for (int kk = 0; kk < D_; kk += 32) {
        // stage Wcat[0..191][kk..kk+31] -> Wl : 768 16B chunks, 3 per thread
#pragma unroll
        for (int c = 0; c < 3; c++) {
            int cid = tid + c * 256;                 // 0..767
            int row = cid >> 2, ch = cid & 3;
            async_copy16((const char*)(Wb + (size_t)row * D_ + kk) + ch * 16,
                         (char*)Wl + row * 80 + ch * 16);
        }
        wait_async();
        __syncthreads();

        Frag a;
        loadA(a, ap + kk, D_, lane);
#pragma unroll
        for (int ct = 0; ct < 12; ct++) {
            Frag w;
            loadB(w, Wl + ct * 16 * 40, 40, lane);
            acc[ct] = wmma_bf16(a, w, acc[ct]);
        }
        __syncthreads();                             // before next stage overwrite
    }

#pragma unroll
    for (int ct = 0; ct < 12; ct++) {
        const int g = ct >> 2, cl = ct & 3;          // 0=q 1=k 2=v
        const float* bias = (g == 0) ? bq : (g == 1) ? bk : bv;
        const float bval = bias[cl * 16 + r];
        v8f a = acc[ct];
#pragma unroll
        for (int i = 0; i < 8; i++) a[i] += bval;

        if (g < 2) {
            __bf16* dst = (g == 0) ? qb : kb;
#pragma unroll
            for (int i = 0; i < 8; i++) {
                int m = i + 8 * h;
                dst[((size_t)(b * N_ + row0 + m)) * DK_ + cl * 16 + r] = (__bf16)a[i];
            }
        } else {
            const int d = cl * 16 + r;
            uint4 st;
            unsigned pw[4];
#pragma unroll
            for (int i = 0; i < 4; i++) {
                union { __bf16 x[2]; unsigned u; } pk;
                pk.x[0] = (__bf16)a[2 * i];
                pk.x[1] = (__bf16)a[2 * i + 1];
                pw[i] = pk.u;
            }
            st.x = pw[0]; st.y = pw[1]; st.z = pw[2]; st.w = pw[3];
            *(uint4*)(vT + ((size_t)(b * DV_ + d)) * N_ + row0 + h * 8) = st;
        }
    }
}

// ---- online softmax update over 64 keys (4 score tiles) + P -> LDS ----
__device__ __forceinline__ void online_update(
    v8f s[4], float (&m)[8], float (&l)[8], v8f (&O)[4],
    __bf16* Pw, int r, int h)
{
#pragma unroll
    for (int i = 0; i < 8; i++) {
        float v0 = s[0][i] * SSCALE, v1 = s[1][i] * SSCALE;
        float v2 = s[2][i] * SSCALE, v3 = s[3][i] * SSCALE;
        float mx = fmaxf(fmaxf(v0, v1), fmaxf(v2, v3));
#pragma unroll
        for (int msk = 1; msk < 16; msk <<= 1) mx = fmaxf(mx, __shfl_xor(mx, msk, 32));
        float mnew = fmaxf(m[i], mx);
        float corr = __expf(m[i] - mnew);
        float e0 = __expf(v0 - mnew), e1 = __expf(v1 - mnew);
        float e2 = __expf(v2 - mnew), e3 = __expf(v3 - mnew);
        float sm = (e0 + e1) + (e2 + e3);
#pragma unroll
        for (int msk = 1; msk < 16; msk <<= 1) sm += __shfl_xor(sm, msk, 32);
        l[i] = l[i] * corr + sm;
        m[i] = mnew;
#pragma unroll
        for (int t = 0; t < 4; t++) O[t][i] *= corr;
        const int mm = i + 8 * h;
        Pw[mm * 72 + r]      = (__bf16)e0;
        Pw[mm * 72 + 16 + r] = (__bf16)e1;
        Pw[mm * 72 + 32 + r] = (__bf16)e2;
        Pw[mm * 72 + 48 + r] = (__bf16)e3;
    }
}

__device__ __forceinline__ void pv_accum(
    v8f (&O)[4], const __bf16* Pw, const __bf16* Vl, int lane)
{
    wait_lds();
#pragma unroll
    for (int kh = 0; kh < 2; kh++) {
        Frag pf;
        loadA(pf, Pw + kh * 32, 72, lane);
#pragma unroll
        for (int t = 0; t < 4; t++) {
            Frag vf;
            loadB(vf, Vl + t * 16 * 72 + kh * 32, 72, lane);
            O[t] = wmma_bf16(pf, vf, O[t]);
        }
    }
}

// ---------------- streaming differential attention, split-K ----------------
// Block: 8 waves = 128 query rows, one key chunk of 1024. K/V tiles (64 keys)
// are async-staged into LDS and shared by all 8 waves. Partial (m,l,O1,O2)
// written to workspace; combine kernel merges the NSPLIT chunks.
__global__ __launch_bounds__(256) void diff_attn(
    const __bf16* __restrict__ qb, const __bf16* __restrict__ kb,
    const __bf16* __restrict__ vT,
    float* __restrict__ Opart, float* __restrict__ MLpart)
{
    __shared__ __align__(16) __bf16 Kl[64 * 72];       // [key][h-dim], stride 72
    __shared__ __align__(16) __bf16 Vl[64 * 72];       // [v-dim][key], stride 72
    __shared__ __align__(16) __bf16 Pl[8][16 * 72];    // per-wave P bounce, stride 72

    const int tid  = threadIdx.x;
    const int lane = tid & 31, warp = tid >> 5;
    const int r = lane & 15, h = lane >> 4;
    const int blk   = blockIdx.x;                      // B*(N/128)*NSPLIT = 256
    const int chunk = blk & (NSPLIT - 1);
    const int rb    = blk >> 2;                        // 0..63
    const int b     = rb / (N_ / 128);
    const int row0  = (rb % (N_ / 128)) * 128 + warp * 16;
    const int j0    = chunk * CHUNK;

    Frag q1, q2;
    const __bf16* qp = qb + ((size_t)(b * N_ + row0)) * DK_;
    loadA(q1, qp,      DK_, lane);
    loadA(q2, qp + 32, DK_, lane);

    v8f O1[4], O2[4];
    v8f vzero = {};
#pragma unroll
    for (int t = 0; t < 4; t++) { O1[t] = vzero; O2[t] = vzero; }
    float m1[8], l1[8], m2[8], l2[8];
#pragma unroll
    for (int i = 0; i < 8; i++) { m1[i] = -1e30f; l1[i] = 0.f; m2[i] = -1e30f; l2[i] = 0.f; }

    const char* kg0 = (const char*)kb + ((size_t)(b * N_)) * DK_ * 2;
    const char* vg0 = (const char*)vT + ((size_t)(b * DV_) * N_) * 2;

    for (int j = j0; j < j0 + CHUNK; j += 64) {
        // stage K[j..j+63][0..63] and V^T[0..63][j..j+63]: 512+512 16B chunks
        const char* kg = kg0 + (size_t)j * 128;        // 64 dims * 2B per key
        const char* vg = vg0 + (size_t)j * 2;          // row stride N_*2
#pragma unroll
        for (int c = 0; c < 2; c++) {
            int cid = tid + c * 256;                   // 0..511
            int row = cid >> 3, ch = cid & 7;
            async_copy16(kg + (size_t)row * 128 + ch * 16,
                         (char*)Kl + row * 144 + ch * 16);
            async_copy16(vg + (size_t)row * (N_ * 2) + ch * 16,
                         (char*)Vl + row * 144 + ch * 16);
        }
        wait_async();
        __syncthreads();

        // ---- half 1: h-dims 0..31 ----
        {
            v8f s[4];
#pragma unroll
            for (int tk = 0; tk < 4; tk++) {
                Frag kf;
                loadB(kf, Kl + tk * 16 * 72, 72, lane);
                s[tk] = wmma_bf16(q1, kf, vzero);
            }
            online_update(s, m1, l1, O1, &Pl[warp][0], r, h);
            pv_accum(O1, &Pl[warp][0], Vl, lane);
        }
        // ---- half 2: h-dims 32..63 ----
        {
            v8f s[4];
#pragma unroll
            for (int tk = 0; tk < 4; tk++) {
                Frag kf;
                loadB(kf, Kl + tk * 16 * 72 + 32, 72, lane);
                s[tk] = wmma_bf16(q2, kf, vzero);
            }
            online_update(s, m2, l2, O2, &Pl[warp][0], r, h);
            pv_accum(O2, &Pl[warp][0], Vl, lane);
        }
        __syncthreads();                               // before next K/V stage
    }

    // ---- write partials ----
    const int tileIdx = b * (N_ / 16) + (row0 >> 4);
    float* Op = Opart + ((size_t)(tileIdx * NSPLIT + chunk)) * 2048;
#pragma unroll
    for (int i = 0; i < 8; i++) {
        const int mm = i + 8 * h;
#pragma unroll
        for (int t = 0; t < 4; t++) {
            Op[mm * 64 + t * 16 + r]        = O1[t][i];
            Op[1024 + mm * 64 + t * 16 + r] = O2[t][i];
        }
    }
    if (r == 0) {
        float4* MLp = (float4*)MLpart + (size_t)(tileIdx * NSPLIT + chunk) * 16;
#pragma unroll
        for (int i = 0; i < 8; i++) {
            const int mm = i + 8 * h;
            float4 ml;
            ml.x = m1[i]; ml.y = l1[i]; ml.z = m2[i]; ml.w = l2[i];
            MLp[mm] = ml;
        }
    }
}

// ---------------- split-K combine: out = O1/L1 - lam*O2/L2 ----------------
__global__ __launch_bounds__(256) void combine_k(
    const float* __restrict__ Opart, const float* __restrict__ MLpart,
    const float* __restrict__ lamp, float* __restrict__ out)
{
    const int tile = blockIdx.x;                       // B*(N/16) = 512
    const int tid  = threadIdx.x;
    const int m  = tid >> 4;                           // 0..15
    const int dB = (tid & 15) * 4;                     // 0..60
    const int b    = tile / (N_ / 16);
    const int row0 = (tile % (N_ / 16)) * 16;
    const float lam = lamp[0];

    const float4* MLp = (const float4*)MLpart + (size_t)tile * NSPLIT * 16;
    float4 ml[NSPLIT];
#pragma unroll
    for (int c = 0; c < NSPLIT; c++) ml[c] = MLp[c * 16 + m];

    float M1 = ml[0].x, M2 = ml[0].z;
#pragma unroll
    for (int c = 1; c < NSPLIT; c++) { M1 = fmaxf(M1, ml[c].x); M2 = fmaxf(M2, ml[c].z); }
    float L1 = 0.f, L2 = 0.f, w1[NSPLIT], w2[NSPLIT];
#pragma unroll
    for (int c = 0; c < NSPLIT; c++) {
        w1[c] = __expf(ml[c].x - M1);
        w2[c] = __expf(ml[c].z - M2);
        L1 += ml[c].y * w1[c];
        L2 += ml[c].w * w2[c];
    }

    float a1x = 0, a1y = 0, a1z = 0, a1w = 0, a2x = 0, a2y = 0, a2z = 0, a2w = 0;
#pragma unroll
    for (int c = 0; c < NSPLIT; c++) {
        const float* Op = Opart + ((size_t)(tile * NSPLIT + c)) * 2048 + m * 64 + dB;
        float4 o1 = *(const float4*)Op;
        float4 o2 = *(const float4*)(Op + 1024);
        a1x += o1.x * w1[c]; a1y += o1.y * w1[c]; a1z += o1.z * w1[c]; a1w += o1.w * w1[c];
        a2x += o2.x * w2[c]; a2y += o2.y * w2[c]; a2z += o2.z * w2[c]; a2w += o2.w * w2[c];
    }
    const float i1 = 1.f / L1, i2 = lam / L2;
    float4 res;
    res.x = a1x * i1 - a2x * i2;
    res.y = a1y * i1 - a2y * i2;
    res.z = a1z * i1 - a2z * i2;
    res.w = a1w * i1 - a2w * i2;
    *(float4*)(out + ((size_t)(b * N_ + row0 + m)) * DV_ + dB) = res;
}

extern "C" void kernel_launch(void* const* d_in, const int* in_sizes, int n_in,
                              void* d_out, int out_size, void* d_ws, size_t ws_size,
                              hipStream_t stream) {
    (void)in_sizes; (void)n_in; (void)out_size; (void)ws_size;
    const float* x   = (const float*)d_in[0];
    const float* Wq  = (const float*)d_in[1];
    const float* bq  = (const float*)d_in[2];
    const float* Wk  = (const float*)d_in[3];
    const float* bk  = (const float*)d_in[4];
    const float* Wv  = (const float*)d_in[5];
    const float* bv  = (const float*)d_in[6];
    const float* lam = (const float*)d_in[7];
    float* out = (float*)d_out;

    // workspace layout (bytes)
    const size_t XB_OFF = 0;                                     // bf16 x: B*N*D
    const size_t WB_OFF = XB_OFF + (size_t)B_ * N_ * D_ * 2;     // bf16 Wcat: 192*D
    const size_t QB_OFF = WB_OFF + (size_t)192 * D_ * 2;         // bf16 q
    const size_t KB_OFF = QB_OFF + (size_t)B_ * N_ * DK_ * 2;    // bf16 k
    const size_t VT_OFF = KB_OFF + (size_t)B_ * N_ * DK_ * 2;    // bf16 vT
    const size_t OP_OFF = VT_OFF + (size_t)B_ * DV_ * N_ * 2;    // f32 partial O
    const size_t ML_OFF = OP_OFF + (size_t)512 * NSPLIT * 2048 * 4; // f32 partial m/l

    char* ws = (char*)d_ws;
    __bf16* xb = (__bf16*)(ws + XB_OFF);
    __bf16* Wb = (__bf16*)(ws + WB_OFF);
    __bf16* qb = (__bf16*)(ws + QB_OFF);
    __bf16* kb = (__bf16*)(ws + KB_OFF);
    __bf16* vT = (__bf16*)(ws + VT_OFF);
    float* Opart  = (float*)(ws + OP_OFF);
    float* MLpart = (float*)(ws + ML_OFF);

    const int xcount = B_ * N_ * D_;
    cvt_f32_bf16<<<(xcount + 255) / 256, 256, 0, stream>>>(x, xb, xcount);
    const int wcount = 64 * D_;
    cvt_f32_bf16<<<(wcount + 255) / 256, 256, 0, stream>>>(Wq, Wb,            wcount);
    cvt_f32_bf16<<<(wcount + 255) / 256, 256, 0, stream>>>(Wk, Wb +  64 * D_, wcount);
    cvt_f32_bf16<<<(wcount + 255) / 256, 256, 0, stream>>>(Wv, Wb + 128 * D_, wcount);

    qkv_proj<<<64, 256, 0, stream>>>(xb, Wb, bq, bk, bv, qb, kb, vT);
    diff_attn<<<B_ * (N_ / 128) * NSPLIT, 256, 0, stream>>>(qb, kb, vT, Opart, MLpart);
    combine_k<<<B_ * (N_ / 16), 256, 0, stream>>>(Opart, MLpart, lam, out);
}
